// Grok5PhiCore_70136815944177
// MI455X (gfx1250) — matrix operand order repaired and checked by reading, not compiled, not run on hardware
//
#include <hip/hip_runtime.h>
#include <hip/hip_bf16.h>
#include <stdint.h>

typedef __bf16 bf16;
typedef __attribute__((ext_vector_type(16))) __bf16 v16bf;
typedef __attribute__((ext_vector_type(8)))  float  v8f;
typedef __attribute__((ext_vector_type(4)))  float  v4f;
typedef __attribute__((ext_vector_type(4)))  unsigned int u32x4;
typedef __attribute__((ext_vector_type(8)))  int    i32x8;
typedef __attribute__((ext_vector_type(4)))  int    i32x4;

#define BDIM 2
#define LDIM 2048
#define DDIM 1024
#define HDIM 16
#define DH   64
#define BL   (BDIM * LDIM)
#define PHI_F 1.6180339887498949f

// ---------------- WMMA fragment helpers (wave32, 16x16x32 bf16) ----------------

__device__ __forceinline__ v16bf frag_from_2x16B(const bf16* p0, const bf16* p1) {
  union { v16bf v; v4f f[2]; } u;
  u.f[0] = *(const v4f*)p0;
  u.f[1] = *(const v4f*)p1;
  return u.v;
}

// A-matrix 16x32 (MxK), row-major source, leading dim lda (elements).
__device__ __forceinline__ v16bf load_A(const bf16* base, int lda) {
  int lane = threadIdx.x & 31;
  int m  = lane & 15;
  int kh = (lane >> 4) << 3;       // 0 or 8
  const bf16* row = base + (size_t)m * lda;
  return frag_from_2x16B(row + kh, row + 16 + kh);
}

// B-matrix 32x16 (KxN), stored column-major-in-N: elem(n,k) at base[n*ldn + k].
__device__ __forceinline__ v16bf load_B(const bf16* base, int ldn) {
  int lane = threadIdx.x & 31;
  int n  = lane & 15;
  int kh = (lane >> 4) << 3;       // 0 or 8
  const bf16* col = base + (size_t)n * ldn;
  return frag_from_2x16B(col + kh, col + 16 + kh);
}

__device__ __forceinline__ v8f wmma_bf16(v16bf a, v16bf b, v8f c) {
  return __builtin_amdgcn_wmma_f32_16x16x32_bf16(false, a, false, b, (short)0, c,
                                                 false, false);
}

// ---------------- TDM: 2D bf16 tile (tile_k x tile_rows) global -> LDS ----------------
// D# per CDNA5 ISA ch.8: group0 = {count/lds/global/type}, group1 = dims/strides.
__device__ __forceinline__ void tdm_load_tile_bf16(uint32_t lds_off, const bf16* gptr,
                                                   uint32_t tile_k, uint32_t tile_rows,
                                                   uint32_t lda, uint32_t tensor_rows) {
  uint64_t ga = (uint64_t)(uintptr_t)gptr;
  u32x4 g0;
  g0[0] = 1u;                                            // count=1 (valid), user mode
  g0[1] = lds_off;                                       // lds_addr [63:32]
  g0[2] = (uint32_t)(ga & 0xFFFFFFFFu);                  // global_addr lo
  g0[3] = (uint32_t)((ga >> 32) & 0x01FFFFFFu) | (2u << 30); // global_addr hi | type=2
  i32x8 g1;
  g1[0] = (int)(1u << 16);                               // wg_mask=0, data_size=1 (2B)
  g1[1] = (int)((lda & 0xFFFFu) << 16);                  // tensor_dim0[15:0] @bit48
  g1[2] = (int)(((lda >> 16) & 0xFFFFu) |
                ((tensor_rows & 0xFFFFu) << 16));        // dim0 hi | tensor_dim1 lo
  g1[3] = (int)(((tensor_rows >> 16) & 0xFFFFu) |
                ((tile_k & 0xFFFFu) << 16));             // dim1 hi | tile_dim0
  g1[4] = (int)(tile_rows & 0xFFFFu);                    // tile_dim1 | tile_dim2=0
  g1[5] = (int)lda;                                      // tensor_dim0_stride[31:0]
  g1[6] = 0;                                             // stride hi | dim1_stride lo
  g1[7] = 0;
  i32x4 z4 = {};
#if __has_include(<hip/amd_detail/amd_gfx1250_TDM.h>)
  i32x8 z8 = {};
  __builtin_amdgcn_tensor_load_to_lds(g0, g1, z4, z4, z8, 0);
#else
  __builtin_amdgcn_tensor_load_to_lds(g0, g1, z4, z4, 0);
#endif
}

// ---------------- conversion / transpose kernels ----------------

__global__ void f32_to_bf16_kernel(const float* __restrict__ src,
                                   bf16* __restrict__ dst, int n) {
  int i = blockIdx.x * blockDim.x + threadIdx.x;
  if (i < n) dst[i] = (bf16)src[i];
}

// src: [Kd][Nd] row-major f32  ->  dst: [Nd][Kd] row-major bf16
__global__ void transpose_to_bf16_kernel(const float* __restrict__ src,
                                         bf16* __restrict__ dst, int Kd, int Nd) {
  int i = blockIdx.x * blockDim.x + threadIdx.x;
  if (i < Kd * Nd) {
    int n = i / Kd;
    int k = i - n * Kd;
    dst[i] = (bf16)src[(size_t)k * Nd + n];
  }
}

// ---------------- QKV GEMM: X[BL,D] @ Wqkv[D,3D] -> scatter to Q/K/V^T ----------------
// A tile (64x64 bf16) staged through LDS by the Tensor Data Mover, double-buffered.

__global__ void __launch_bounds__(256)
qkv_gemm_kernel(const bf16* __restrict__ Xbf, const bf16* __restrict__ WqkvT,
                bf16* __restrict__ Q, bf16* __restrict__ K, bf16* __restrict__ VT) {
  __shared__ __align__(16) bf16 atile[2][64 * 64];       // 16 KiB double buffer

  int tileM0 = blockIdx.x * 64;
  int tileN0 = blockIdx.y * 64;
  int wave = threadIdx.x >> 5;
  int lane = threadIdx.x & 31;
  int ti = wave >> 1;
  int tj = (wave & 1) * 2;
  int m0l = ti * 16;                                     // row offset inside LDS tile
  int n0a = tileN0 + tj * 16;
  int n0b = n0a + 16;

  v8f acc0 = {}, acc1 = {};
  const bf16* Ag     = Xbf + (size_t)tileM0 * DDIM;
  const bf16* B0base = WqkvT + (size_t)n0a * DDIM;
  const bf16* B1base = WqkvT + (size_t)n0b * DDIM;

  if (threadIdx.x < 32)
    tdm_load_tile_bf16((uint32_t)(uintptr_t)&atile[0][0], Ag, 64, 64, DDIM, BL);

  for (int k0 = 0; k0 < DDIM; k0 += 64) {
    int cur = (k0 >> 6) & 1;
    if (threadIdx.x < 32) {
      if (k0 + 64 < DDIM) {
        tdm_load_tile_bf16((uint32_t)(uintptr_t)&atile[cur ^ 1][0], Ag + k0 + 64,
                           64, 64, DDIM, BL);
        __builtin_amdgcn_s_wait_tensorcnt(1);            // oldest (cur) complete
      } else {
        __builtin_amdgcn_s_wait_tensorcnt(0);
      }
    }
    __syncthreads();
    const bf16* As = &atile[cur][m0l * 64];
#pragma unroll
    for (int kk = 0; kk < 64; kk += 32) {
      __builtin_prefetch(B0base + k0 + kk + 128, 0, 0);
      v16bf a  = load_A(As + kk, 64);                    // ds_load_b128 pairs
      v16bf b0 = load_B(B0base + k0 + kk, DDIM);
      v16bf b1 = load_B(B1base + k0 + kk, DDIM);
      acc0 = wmma_bf16(a, b0, acc0);
      acc1 = wmma_bf16(a, b1, acc1);
    }
    __syncthreads();                                     // before buffer reuse
  }

  int colL  = lane & 15;
  int rbase = (lane >> 4) << 3;
#pragma unroll
  for (int r = 0; r < 8; ++r) {
    int row = tileM0 + m0l + rbase + r;                  // [0, BL)
    int b = row >> 11;
    int l = row & 2047;
#pragma unroll
    for (int s = 0; s < 2; ++s) {
      int n = (s == 0 ? n0a : n0b) + colL;
      float v = (s == 0) ? acc0[r] : acc1[r];
      int sel = n >> 10;                                 // 0=Q 1=K 2=V
      int nn  = n & 1023;
      int h   = nn >> 6;
      int dh  = nn & 63;
      bf16 bv = (bf16)v;
      size_t bh = (size_t)(b * HDIM + h);
      if (sel == 0)      Q[(bh * LDIM + l) * DH + dh] = bv;
      else if (sel == 1) K[(bh * LDIM + l) * DH + dh] = bv;
      else               VT[(bh * DH + dh) * LDIM + l] = bv;
    }
  }
}

// ---------------- Flash attention: per wave 16 query rows, chunks of 64 keys ----------------

__global__ void __launch_bounds__(128)
attn_kernel(const bf16* __restrict__ Q, const bf16* __restrict__ Kb,
            const bf16* __restrict__ VT, bf16* __restrict__ O) {
  __shared__ __align__(16) bf16 pbuf[4][16 * 64];        // per-wave P staging

  int bh = blockIdx.x >> 5;                              // 32 row-tiles per (b,h)
  int rt = blockIdx.x & 31;
  int b = bh >> 4;
  int h = bh & 15;
  int wave = threadIdx.x >> 5;
  int lane = threadIdx.x & 31;
  int row0 = rt * 64 + wave * 16;

  const bf16* qbase  = Q  + ((size_t)bh * LDIM + row0) * DH;
  const bf16* kbase  = Kb + (size_t)bh * LDIM * DH;
  const bf16* vtbase = VT + (size_t)bh * DH * LDIM;
  bf16* myp = &pbuf[wave][0];

  v16bf qa0 = load_A(qbase, DH);                         // K-dim 0..31
  v16bf qa1 = load_A(qbase + 32, DH);                    // K-dim 32..63

  int colL  = lane & 15;
  int rbase = (lane >> 4) << 3;
  const float scale = 0.125f;                            // 64^-0.5

  float mstate[8], lstate[8], posi[8];
  v8f acc[4] = {};
#pragma unroll
  for (int r = 0; r < 8; ++r) {
    mstate[r] = -1e30f;
    lstate[r] = 0.0f;
    int i = row0 + rbase + r;
    posi[r] = fmodf((float)i * PHI_F, 1.0f);
  }

  for (int j0 = 0; j0 < LDIM; j0 += 64) {
    // ---- S = Q K^T for 16x64 chunk ----
    v8f s[4];
#pragma unroll
    for (int sub = 0; sub < 4; ++sub) {
      int n0 = j0 + sub * 16;
      v8f sv = {};
      sv = wmma_bf16(qa0, load_B(kbase + (size_t)n0 * DH, DH), sv);
      sv = wmma_bf16(qa1, load_B(kbase + (size_t)n0 * DH + 32, DH), sv);
      s[sub] = sv;
    }

    // ---- scale + phi bias, per-row max ----
    float rmax[8];
#pragma unroll
    for (int r = 0; r < 8; ++r) rmax[r] = -1e30f;
#pragma unroll
    for (int sub = 0; sub < 4; ++sub) {
      int j = j0 + sub * 16 + colL;
      float posj = fmodf((float)j * PHI_F, 1.0f);
#pragma unroll
      for (int r = 0; r < 8; ++r) {
        float d = fabsf(posi[r] - posj);
        float bias = __logf(__expf(-d) + 1e-8f);
        float val = s[sub][r] * scale + bias;
        s[sub][r] = val;
        rmax[r] = fmaxf(rmax[r], val);
      }
    }
#pragma unroll
    for (int r = 0; r < 8; ++r) {
      float v = rmax[r];
      v = fmaxf(v, __shfl_xor(v, 1, 32));
      v = fmaxf(v, __shfl_xor(v, 2, 32));
      v = fmaxf(v, __shfl_xor(v, 4, 32));
      v = fmaxf(v, __shfl_xor(v, 8, 32));
      rmax[r] = v;
    }

    float alpha[8], mnew[8], rsum[8];
#pragma unroll
    for (int r = 0; r < 8; ++r) {
      mnew[r]  = fmaxf(mstate[r], rmax[r]);
      alpha[r] = __expf(mstate[r] - mnew[r]);
      mstate[r] = mnew[r];
      rsum[r] = 0.0f;
    }

    // ---- P = exp(S - m) -> LDS for A-fragment reload ----
#pragma unroll
    for (int sub = 0; sub < 4; ++sub) {
#pragma unroll
      for (int r = 0; r < 8; ++r) {
        float p = __expf(s[sub][r] - mnew[r]);
        rsum[r] += p;
        myp[(rbase + r) * 64 + sub * 16 + colL] = (bf16)p;
      }
    }
#pragma unroll
    for (int r = 0; r < 8; ++r) {
      float v = rsum[r];
      v += __shfl_xor(v, 1, 32);
      v += __shfl_xor(v, 2, 32);
      v += __shfl_xor(v, 4, 32);
      v += __shfl_xor(v, 8, 32);
      lstate[r] = lstate[r] * alpha[r] + v;
    }

#pragma unroll
    for (int sub = 0; sub < 4; ++sub)
#pragma unroll
      for (int r = 0; r < 8; ++r) acc[sub][r] *= alpha[r];

    asm volatile("s_wait_dscnt 0x0" ::: "memory");

    // ---- O += P V ----
    v16bf pa0 = load_A(myp, 64);
    v16bf pa1 = load_A(myp + 32, 64);
#pragma unroll
    for (int sub = 0; sub < 4; ++sub) {
      int n0 = sub * 16;
      const bf16* vb = vtbase + (size_t)n0 * LDIM + j0;
      acc[sub] = wmma_bf16(pa0, load_B(vb, LDIM), acc[sub]);
      acc[sub] = wmma_bf16(pa1, load_B(vb + 32, LDIM), acc[sub]);
    }
  }

#pragma unroll
  for (int r = 0; r < 8; ++r) {
    float inv = 1.0f / lstate[r];
    int row = row0 + rbase + r;
    size_t orow = ((size_t)b * LDIM + row) * DDIM + (size_t)h * DH;
#pragma unroll
    for (int sub = 0; sub < 4; ++sub)
      O[orow + sub * 16 + colL] = (bf16)(acc[sub][r] * inv);
  }
}

// ---------------- Output projection: O[BL,D] @ Wproj[D,D] + bias -> f32 ----------------

__global__ void __launch_bounds__(256)
proj_gemm_kernel(const bf16* __restrict__ Obf, const bf16* __restrict__ WprojT,
                 const float* __restrict__ bias, float* __restrict__ out) {
  __shared__ __align__(16) bf16 atile[2][64 * 64];

  int tileM0 = blockIdx.x * 64;
  int tileN0 = blockIdx.y * 64;
  int wave = threadIdx.x >> 5;
  int lane = threadIdx.x & 31;
  int ti = wave >> 1;
  int tj = (wave & 1) * 2;
  int m0l = ti * 16;
  int n0a = tileN0 + tj * 16;
  int n0b = n0a + 16;

  v8f acc0 = {}, acc1 = {};
  const bf16* Ag     = Obf + (size_t)tileM0 * DDIM;
  const bf16* B0base = WprojT + (size_t)n0a * DDIM;
  const bf16* B1base = WprojT + (size_t)n0b * DDIM;

  if (threadIdx.x < 32)
    tdm_load_tile_bf16((uint32_t)(uintptr_t)&atile[0][0], Ag, 64, 64, DDIM, BL);

  for (int k0 = 0; k0 < DDIM; k0 += 64) {
    int cur = (k0 >> 6) & 1;
    if (threadIdx.x < 32) {
      if (k0 + 64 < DDIM) {
        tdm_load_tile_bf16((uint32_t)(uintptr_t)&atile[cur ^ 1][0], Ag + k0 + 64,
                           64, 64, DDIM, BL);
        __builtin_amdgcn_s_wait_tensorcnt(1);
      } else {
        __builtin_amdgcn_s_wait_tensorcnt(0);
      }
    }
    __syncthreads();
    const bf16* As = &atile[cur][m0l * 64];
#pragma unroll
    for (int kk = 0; kk < 64; kk += 32) {
      __builtin_prefetch(B0base + k0 + kk + 128, 0, 0);
      v16bf a  = load_A(As + kk, 64);
      v16bf b0 = load_B(B0base + k0 + kk, DDIM);
      v16bf b1 = load_B(B1base + k0 + kk, DDIM);
      acc0 = wmma_bf16(a, b0, acc0);
      acc1 = wmma_bf16(a, b1, acc1);
    }
    __syncthreads();
  }

  int colL  = lane & 15;
  int rbase = (lane >> 4) << 3;
#pragma unroll
  for (int r = 0; r < 8; ++r) {
    int row = tileM0 + m0l + rbase + r;
    out[(size_t)row * DDIM + n0a + colL] = acc0[r] + bias[n0a + colL];
    out[(size_t)row * DDIM + n0b + colL] = acc1[r] + bias[n0b + colL];
  }
}

// ---------------- host launcher ----------------

extern "C" void kernel_launch(void* const* d_in, const int* in_sizes, int n_in,
                              void* d_out, int out_size, void* d_ws, size_t ws_size,
                              hipStream_t stream) {
  (void)in_sizes; (void)n_in; (void)out_size; (void)ws_size;
  const float* x      = (const float*)d_in[0];   // [2,2048,1024]
  const float* w_qkv  = (const float*)d_in[1];   // [1024,3072]
  const float* w_proj = (const float*)d_in[2];   // [1024,1024]
  const float* b_proj = (const float*)d_in[3];   // [1024]
  float* out = (float*)d_out;

  char* ws = (char*)d_ws;
  bf16* Xbf    = (bf16*)ws; ws += (size_t)BL * DDIM * sizeof(bf16);               // 8 MiB
  bf16* WqkvT  = (bf16*)ws; ws += (size_t)3 * DDIM * DDIM * sizeof(bf16);         // 6 MiB
  bf16* WprojT = (bf16*)ws; ws += (size_t)DDIM * DDIM * sizeof(bf16);             // 2 MiB
  bf16* Qb     = (bf16*)ws; ws += (size_t)BDIM * HDIM * LDIM * DH * sizeof(bf16); // 8 MiB
  bf16* Kb     = (bf16*)ws; ws += (size_t)BDIM * HDIM * LDIM * DH * sizeof(bf16); // 8 MiB
  bf16* VTb    = (bf16*)ws; ws += (size_t)BDIM * HDIM * DH * LDIM * sizeof(bf16); // 8 MiB
  bf16* Ob     = (bf16*)ws;                                                       // 8 MiB

  int nx = BL * DDIM;
  f32_to_bf16_kernel<<<(nx + 255) / 256, 256, 0, stream>>>(x, Xbf, nx);
  int nw1 = 3 * DDIM * DDIM;
  transpose_to_bf16_kernel<<<(nw1 + 255) / 256, 256, 0, stream>>>(w_qkv, WqkvT, DDIM, 3 * DDIM);
  int nw2 = DDIM * DDIM;
  transpose_to_bf16_kernel<<<(nw2 + 255) / 256, 256, 0, stream>>>(w_proj, WprojT, DDIM, DDIM);

  dim3 g1(BL / 64, 3 * DDIM / 64);
  qkv_gemm_kernel<<<g1, 256, 0, stream>>>(Xbf, WqkvT, Qb, Kb, VTb);

  attn_kernel<<<BDIM * HDIM * (LDIM / 64), 128, 0, stream>>>(Qb, Kb, VTb, Ob);

  dim3 g2(BL / 64, DDIM / 64);
  proj_gemm_kernel<<<g2, 256, 0, stream>>>(Ob, WprojT, b_proj, out);
}
